// SpectrogramEncoder_28604482191416
// MI455X (gfx1250) — matrix operand order repaired and checked by reading, not compiled, not run on hardware
//
#include <hip/hip_runtime.h>
#include <hip/hip_bf16.h>
#include <math.h>

// ---------------------------------------------------------------------------
// Griffin-Lim vocoder on gfx1250: DFT-as-GEMM via v_wmma_f32_16x16x32_f16.
// GEMM kernels use 2x2 register macro-tiles (32x32 C per wave) so the K-loop
// issues 8 WMMAs per 6-8 fragment loads instead of 2 per 3-4.
// ---------------------------------------------------------------------------
//  B=8, GH=32, GW=64, FULL=128x512, N_FFT=1024, HOP=256, N_FREQ=513
//  frames M = 8*512 = 4096, freq padded FP = 544 (=17*32 = 34*16), time NF=1024
//  L (signal) = HOP*(W-1) = 130816, padded OLA length = 131840, out = 131072
// ---------------------------------------------------------------------------

typedef __attribute__((ext_vector_type(16))) _Float16 v16h;
typedef __attribute__((ext_vector_type(8)))  _Float16 v8h;
typedef __attribute__((ext_vector_type(8)))  float    v8f;

#define GL_M      4096
#define GL_FP     544
#define GL_NFREQ  513
#define GL_NF     1024
#define GL_L      130816
#define GL_SIG    131840
#define GL_NS     131072
#define GL_B      8
#define GL_PI     3.14159265358979323846f

#define GL_WMMA(a, b, c) \
  __builtin_amdgcn_wmma_f32_16x16x32_f16(false, (a), false, (b), (short)0, (c), false, false)

// ------------------------------ WMMA fragment helpers ----------------------
// A matrix 16x32 f16, row-major source, per ISA layout:
//   lane r=lane&15 holds row r; lanes<16 -> K groups {0..7,16..23},
//   lanes>=16 -> {8..15,24..31}. Elements contiguous in K => two b128 loads.
__device__ __forceinline__ v16h gl_load_a(const _Float16* A, int ld, int row0,
                                          int k0, int lane) {
  int r  = lane & 15;
  int kb = (lane & 16) ? 8 : 0;
  const _Float16* p = A + (size_t)(row0 + r) * ld + k0 + kb;
  union { v16h v; v8h h[2]; } u;
  u.h[0] = *(const v8h*)(p);
  u.h[1] = *(const v8h*)(p + 16);
  return u.v;
}

// B matrix 32x16 f16 from TRANSPOSED (N-major) storage BT[N][K], ld = K dim.
//   lane c=lane&15 holds column c; lanes<16 -> K {0..7,16..23}, lanes>=16 -> {8..15,24..31}.
__device__ __forceinline__ v16h gl_load_bT(const _Float16* BT, int ld, int k0,
                                           int col0, int lane) {
  int c    = lane & 15;
  int koff = (lane & 16) ? 8 : 0;
  const _Float16* p = BT + (size_t)(col0 + c) * ld + k0 + koff;
  union { v16h v; v8h h[2]; } u;
  u.h[0] = *(const v8h*)(p);
  u.h[1] = *(const v8h*)(p + 16);
  return u.v;
}

// C/D 16x16 f32: VGPR r holds M=r (lanes<16) or M=8+r (lanes>=16), N=lane&15.
__device__ __forceinline__ void gl_store_c(float* D, int ld, int row0, int col0,
                                           int lane, v8f acc) {
  int c  = lane & 15;
  int rb = (lane & 16) ? 8 : 0;
  float* p = D + (size_t)(row0 + rb) * ld + col0 + c;
#pragma unroll
  for (int r = 0; r < 8; ++r) p[(size_t)r * ld] = acc[r];
}

__device__ __forceinline__ float gl_win(int n) {
  return 0.5f - 0.5f * __cosf(2.0f * GL_PI * (float)n * (1.0f / 1024.0f));
}

// ------------------------------ basis generation ---------------------------
// Forward rDFT bases, transposed storage: BcT/BsT shape (FP=544, NF=1024).
//   BcT[k][n] =  cos(2*pi*k*n/1024)   (k < 513 else 0)
//   BsT[k][n] = -sin(2*pi*k*n/1024)
__global__ void gl_fill_fwd_basis(_Float16* BcT, _Float16* BsT) {
  int idx = blockIdx.x * blockDim.x + threadIdx.x;
  if (idx >= GL_FP * GL_NF) return;
  int k = idx / GL_NF, n = idx % GL_NF;
  float c = 0.f, s = 0.f;
  if (k < GL_NFREQ) {
    int w = (k * n) & 1023;                 // period-1024 exact phase
    float ang = (float)w * (2.0f * GL_PI / 1024.0f);
    c = cosf(ang); s = -sinf(ang);
  }
  BcT[idx] = (_Float16)c;
  BsT[idx] = (_Float16)s;
}

// Inverse (irfft) bases with Hermitian weights folded in, transposed storage:
// CtT/StT shape (NF=1024, FP=544):
//   CtT[n][k] =  s_k/1024 * cos(2*pi*k*n/1024),  s_k = (k==0||k==512)?1:2
//   StT[n][k] = -s_k/1024 * sin(2*pi*k*n/1024)
__global__ void gl_fill_inv_basis(_Float16* CtT, _Float16* StT) {
  int idx = blockIdx.x * blockDim.x + threadIdx.x;
  if (idx >= GL_NF * GL_FP) return;
  int n = idx / GL_FP, k = idx % GL_FP;
  float c = 0.f, s = 0.f;
  if (k < GL_NFREQ) {
    float sk = (k == 0 || k == 512) ? 1.0f : 2.0f;
    float sc = sk * (1.0f / 1024.0f);
    int w = (k * n) & 1023;
    float ang = (float)w * (2.0f * GL_PI / 1024.0f);
    c = sc * cosf(ang); s = -sc * sinf(ang);
  }
  CtT[idx] = (_Float16)c;
  StT[idx] = (_Float16)s;
}

// 1/sum(window^2) for OLA normalization over the padded 131840-sample lattice.
__global__ void gl_fill_wsqinv(float* wsqInv) {
  int i = blockIdx.x * blockDim.x + threadIdx.x;
  if (i >= GL_SIG) return;
  int fhi = i >> 8; if (fhi > 511) fhi = 511;
  int t = i - 768;
  int flo = (t > 0) ? (t >> 8) : 0;
  float acc = 0.f;
  for (int f = flo; f <= fhi; ++f) {
    float w = gl_win(i - (f << 8));
    acc += w * w;
  }
  wsqInv[i] = (acc > 1e-11f) ? (1.0f / acc) : 1.0f;
}

// ------------------------------ spec prep ----------------------------------
// Bilinear 32x64 -> 128x512 (half-pixel centers), written straight to d_out.
__global__ void gl_resize_full(const float* params, float* full_spec) {
  int idx = blockIdx.x * blockDim.x + threadIdx.x;
  if (idx >= GL_B * 128 * 512) return;
  int x = idx & 511, y = (idx >> 9) & 127, b = idx >> 16;
  float sy = (y + 0.5f) * 0.25f  - 0.5f;    // 32/128
  float sx = (x + 0.5f) * 0.125f - 0.5f;    // 64/512
  int y0 = (int)floorf(sy), x0 = (int)floorf(sx);
  float fy = sy - y0, fx = sx - x0;
  int y1 = y0 + 1, x1 = x0 + 1;
  y0 = min(max(y0, 0), 31); y1 = min(max(y1, 0), 31);
  x0 = min(max(x0, 0), 63); x1 = min(max(x1, 0), 63);
  const float* g = params + (size_t)b * 2048;
  float v00 = g[y0 * 64 + x0], v01 = g[y0 * 64 + x1];
  float v10 = g[y1 * 64 + x0], v11 = g[y1 * 64 + x1];
  float v0 = v00 + (v01 - v00) * fx;
  float v1 = v10 + (v11 - v10) * fx;
  full_spec[idx] = v0 + (v1 - v0) * fy;
}

// mag[(b*512+t)*544 + k] = sqrt(max(resize_H(full_spec^2 * 100), 0))
__global__ void gl_make_mag(const float* full_spec, float* mag) {
  int idx = blockIdx.x * blockDim.x + threadIdx.x;
  if (idx >= GL_M * GL_FP) return;
  int k = idx % GL_FP, m = idx / GL_FP;
  int t = m & 511, b = m >> 9;
  float v = 0.f;
  if (k < GL_NFREQ) {
    float sy = (k + 0.5f) * (128.0f / 513.0f) - 0.5f;
    int y0 = (int)floorf(sy);
    float fy = sy - y0;
    int y1 = y0 + 1;
    y0 = min(max(y0, 0), 127); y1 = min(max(y1, 0), 127);
    float a = full_spec[((size_t)b * 128 + y0) * 512 + t];
    float c = full_spec[((size_t)b * 128 + y1) * 512 + t];
    float p0 = a * a * 100.0f, p1 = c * c * 100.0f;
    float p = p0 + (p1 - p0) * fy;
    v = sqrtf(fmaxf(p, 0.0f));
  }
  mag[idx] = v;
}

// Random initial phase (deterministic hash), S = mag * e^{i*2*pi*u}, tprev=0.
__global__ void gl_init_angles(const float* mag, _Float16* Sre, _Float16* Sim,
                               float* tre, float* tim) {
  int idx = blockIdx.x * blockDim.x + threadIdx.x;
  if (idx >= GL_M * GL_FP) return;
  int k = idx % GL_FP;
  tre[idx] = 0.f; tim[idx] = 0.f;
  float sr = 0.f, si = 0.f;
  if (k < GL_NFREQ) {
    unsigned h = (unsigned)idx * 2654435761u + 0x9E3779B9u;
    h ^= h >> 16; h *= 0x7feb352du; h ^= h >> 15; h *= 0x846ca68bu; h ^= h >> 16;
    float u = (float)(h >> 8) * (1.0f / 16777216.0f);
    float th = 2.0f * GL_PI * u;
    float m = mag[idx];
    sr = m * cosf(th); si = m * sinf(th);
  }
  Sre[idx] = (_Float16)sr;
  Sim[idx] = (_Float16)si;
}

// ------------------------------ WMMA GEMM kernels --------------------------
// ISTFT: T(4096x1024,f32) = Sre(4096x544) x Ct + Sim(4096x544) x St
// 2x2 macro-tile: 8 WMMAs per K-step on 8 fragment loads.
__global__ void gl_istft_gemm(const _Float16* __restrict__ Sre,
                              const _Float16* __restrict__ Sim,
                              const _Float16* __restrict__ CtT,
                              const _Float16* __restrict__ StT,
                              float* __restrict__ T) {
  int wave = (blockIdx.x * blockDim.x + threadIdx.x) >> 5;
  int lane = threadIdx.x & 31;
  const int NT = GL_NF / 32;                // 32 macro column tiles
  int mt = wave / NT, nt = wave % NT;
  if (mt >= GL_M / 32) return;
  int row0 = mt * 32, col0 = nt * 32;
  v8f c00 = {}, c01 = {}, c10 = {}, c11 = {};
#pragma unroll 1
  for (int k0 = 0; k0 < GL_FP; k0 += 32) {  // 17 steps, padded K zeros
    v16h ar0 = gl_load_a (Sre, GL_FP, row0,      k0, lane);
    v16h ar1 = gl_load_a (Sre, GL_FP, row0 + 16, k0, lane);
    v16h ai0 = gl_load_a (Sim, GL_FP, row0,      k0, lane);
    v16h ai1 = gl_load_a (Sim, GL_FP, row0 + 16, k0, lane);
    v16h bc0 = gl_load_bT(CtT, GL_FP, k0, col0,      lane);
    v16h bc1 = gl_load_bT(CtT, GL_FP, k0, col0 + 16, lane);
    v16h bs0 = gl_load_bT(StT, GL_FP, k0, col0,      lane);
    v16h bs1 = gl_load_bT(StT, GL_FP, k0, col0 + 16, lane);
    c00 = GL_WMMA(ar0, bc0, c00);
    c01 = GL_WMMA(ar0, bc1, c01);
    c10 = GL_WMMA(ar1, bc0, c10);
    c11 = GL_WMMA(ar1, bc1, c11);
    c00 = GL_WMMA(ai0, bs0, c00);
    c01 = GL_WMMA(ai0, bs1, c01);
    c10 = GL_WMMA(ai1, bs0, c10);
    c11 = GL_WMMA(ai1, bs1, c11);
  }
  gl_store_c(T, GL_NF, row0,      col0,      lane, c00);
  gl_store_c(T, GL_NF, row0,      col0 + 16, lane, c01);
  gl_store_c(T, GL_NF, row0 + 16, col0,      lane, c10);
  gl_store_c(T, GL_NF, row0 + 16, col0 + 16, lane, c11);
}

// STFT: Re/Im(4096x544,f32) = Frames(4096x1024,f16) x {Bc,Bs}
// 2x2 macro-tile with A shared across cos/sin: 8 WMMAs per 6 fragment loads.
__global__ void gl_stft_gemm(const _Float16* __restrict__ A,
                             const _Float16* __restrict__ BcT,
                             const _Float16* __restrict__ BsT,
                             float* __restrict__ Re, float* __restrict__ Im) {
  int wave = (blockIdx.x * blockDim.x + threadIdx.x) >> 5;
  int lane = threadIdx.x & 31;
  const int NT = GL_FP / 32;                // 17 macro column tiles
  int mt = wave / NT, nt = wave % NT;
  if (mt >= GL_M / 32) return;
  int row0 = mt * 32, col0 = nt * 32;
  v8f r00 = {}, r01 = {}, r10 = {}, r11 = {};
  v8f i00 = {}, i01 = {}, i10 = {}, i11 = {};
#pragma unroll 1
  for (int k0 = 0; k0 < GL_NF; k0 += 32) {  // 32 steps
    v16h a0  = gl_load_a (A,   GL_NF, row0,      k0, lane);
    v16h a1  = gl_load_a (A,   GL_NF, row0 + 16, k0, lane);
    v16h bc0 = gl_load_bT(BcT, GL_NF, k0, col0,      lane);
    v16h bc1 = gl_load_bT(BcT, GL_NF, k0, col0 + 16, lane);
    v16h bs0 = gl_load_bT(BsT, GL_NF, k0, col0,      lane);
    v16h bs1 = gl_load_bT(BsT, GL_NF, k0, col0 + 16, lane);
    r00 = GL_WMMA(a0, bc0, r00);
    r01 = GL_WMMA(a0, bc1, r01);
    r10 = GL_WMMA(a1, bc0, r10);
    r11 = GL_WMMA(a1, bc1, r11);
    i00 = GL_WMMA(a0, bs0, i00);
    i01 = GL_WMMA(a0, bs1, i01);
    i10 = GL_WMMA(a1, bs0, i10);
    i11 = GL_WMMA(a1, bs1, i11);
  }
  gl_store_c(Re, GL_FP, row0,      col0,      lane, r00);
  gl_store_c(Re, GL_FP, row0,      col0 + 16, lane, r01);
  gl_store_c(Re, GL_FP, row0 + 16, col0,      lane, r10);
  gl_store_c(Re, GL_FP, row0 + 16, col0 + 16, lane, r11);
  gl_store_c(Im, GL_FP, row0,      col0,      lane, i00);
  gl_store_c(Im, GL_FP, row0,      col0 + 16, lane, i01);
  gl_store_c(Im, GL_FP, row0 + 16, col0,      lane, i10);
  gl_store_c(Im, GL_FP, row0 + 16, col0 + 16, lane, i11);
}

// ------------------------------ elementwise --------------------------------
// Deterministic gather overlap-add (<=4 fixed-order terms), window + 1/wsq,
// slices off the N_FFT/2 pad: xsig[b][q] for q in [0, L).
__global__ void gl_ola(const float* __restrict__ T,
                       const float* __restrict__ wsqInv,
                       float* __restrict__ xsig) {
  int idx = blockIdx.x * blockDim.x + threadIdx.x;
  if (idx >= GL_B * GL_L) return;
  int q = idx % GL_L, b = idx / GL_L;
  int i = q + 512;
  int fhi = i >> 8; if (fhi > 511) fhi = 511;
  int t = i - 768;
  int flo = (t > 0) ? (t >> 8) : 0;
  float acc = 0.f;
  for (int f = flo; f <= fhi; ++f) {
    int n = i - (f << 8);
    acc += T[((size_t)(b * 512 + f) << 10) + n] * gl_win(n);
  }
  xsig[idx] = acc * wsqInv[i];
}

// Reflect-pad + window + f16 framing for the STFT GEMM A matrix.
__global__ void gl_frame(const float* __restrict__ xsig,
                         _Float16* __restrict__ F) {
  int idx = blockIdx.x * blockDim.x + threadIdx.x;
  if (idx >= GL_M * GL_NF) return;
  int n = idx & 1023, m = idx >> 10;
  int f = m & 511, b = m >> 9;
  int p = (f << 8) + n - 512;               // position in unpadded signal
  if (p < 0) p = -p;
  if (p >= GL_L) p = 2 * GL_L - 2 - p;      // numpy 'reflect'
  F[idx] = (_Float16)(xsig[(size_t)b * GL_L + p] * gl_win(n));
}

// Griffin-Lim phase update: ang = rebuilt - tprev*(M/(1+M)); normalize;
// tprev = rebuilt; S = mag * ang  (f16 operands for next ISTFT GEMM).
__global__ void gl_update(const float* __restrict__ Re,
                          const float* __restrict__ Im,
                          const float* __restrict__ mag,
                          float* __restrict__ tre, float* __restrict__ tim,
                          _Float16* __restrict__ Sre, _Float16* __restrict__ Sim) {
  int idx = blockIdx.x * blockDim.x + threadIdx.x;
  if (idx >= GL_M * GL_FP) return;
  int k = idx % GL_FP;
  if (k >= GL_NFREQ) {
    Sre[idx] = (_Float16)0.f; Sim[idx] = (_Float16)0.f;
    return;
  }
  const float c = 0.99f / 1.99f;            // MOMENTUM/(1+MOMENTUM)
  float re = Re[idx], im = Im[idx];
  float ar = re - tre[idx] * c;
  float ai = im - tim[idx] * c;
  float inv = 1.0f / (sqrtf(ar * ar + ai * ai) + 1e-16f);
  tre[idx] = re; tim[idx] = im;
  float m = mag[idx];
  Sre[idx] = (_Float16)(m * ar * inv);
  Sim[idx] = (_Float16)(m * ai * inv);
}

__global__ void gl_maxabs(const float* __restrict__ xsig, float* __restrict__ mv) {
  __shared__ float red[256];
  int b = blockIdx.x;
  float m = 0.f;
  for (int q = threadIdx.x; q < GL_L; q += 256)
    m = fmaxf(m, fabsf(xsig[(size_t)b * GL_L + q]));
  red[threadIdx.x] = m;
  __syncthreads();
  for (int s = 128; s > 0; s >>= 1) {
    if (threadIdx.x < s) red[threadIdx.x] = fmaxf(red[threadIdx.x], red[threadIdx.x + s]);
    __syncthreads();
  }
  if (threadIdx.x == 0) mv[b] = red[0];
}

__global__ void gl_scale_out(const float* __restrict__ xsig,
                             const float* __restrict__ mv,
                             float* __restrict__ audio) {
  int idx = blockIdx.x * blockDim.x + threadIdx.x;
  if (idx >= GL_B * GL_NS) return;
  int q = idx & (GL_NS - 1), b = idx >> 17;
  float s = 0.9f / fmaxf(mv[b], 1e-8f);
  audio[idx] = (q < GL_L) ? xsig[(size_t)b * GL_L + q] * s : 0.0f;
}

// ------------------------------ host side ----------------------------------
extern "C" void kernel_launch(void* const* d_in, const int* in_sizes, int n_in,
                              void* d_out, int out_size, void* d_ws, size_t ws_size,
                              hipStream_t stream) {
  (void)in_sizes; (void)n_in; (void)out_size; (void)ws_size;
  const float* params = (const float*)d_in[0];
  float* out = (float*)d_out;
  float* audio_out = out;                                // 8 * 131072
  float* full_spec = out + (size_t)GL_B * GL_NS;         // 8 * 128 * 512

  // workspace carve-up (256B-aligned)
  char* ws = (char*)d_ws;
  size_t off = 0;
  auto carve = [&](size_t bytes) -> char* {
    char* p = ws + off;
    off = (off + bytes + 255) & ~(size_t)255;
    return p;
  };
  float*    mag    = (float*)   carve((size_t)GL_M * GL_FP * 4);
  _Float16* Sre    = (_Float16*)carve((size_t)GL_M * GL_FP * 2);
  _Float16* Sim    = (_Float16*)carve((size_t)GL_M * GL_FP * 2);
  float*    tre    = (float*)   carve((size_t)GL_M * GL_FP * 4);
  float*    tim    = (float*)   carve((size_t)GL_M * GL_FP * 4);
  float*    Re     = (float*)   carve((size_t)GL_M * GL_FP * 4);
  float*    Im     = (float*)   carve((size_t)GL_M * GL_FP * 4);
  float*    ft     = (float*)   carve((size_t)GL_M * GL_NF * 4); // istft frames
  _Float16* ff     = (_Float16*)carve((size_t)GL_M * GL_NF * 2); // stft frames
  float*    xsig   = (float*)   carve((size_t)GL_B * GL_L * 4);
  float*    wsqInv = (float*)   carve((size_t)GL_SIG * 4);
  _Float16* BcT    = (_Float16*)carve((size_t)GL_FP * GL_NF * 2);
  _Float16* BsT    = (_Float16*)carve((size_t)GL_FP * GL_NF * 2);
  _Float16* CtT    = (_Float16*)carve((size_t)GL_NF * GL_FP * 2);
  _Float16* StT    = (_Float16*)carve((size_t)GL_NF * GL_FP * 2);
  float*    mv     = (float*)   carve(GL_B * 4);

  const int TB = 256;
  const int nSpec   = GL_M * GL_FP;          // 2,228,224
  const int nFrames = GL_M * GL_NF;          // 4,194,304
  const int gSpec   = (nSpec   + TB - 1) / TB;
  const int gFrames = (nFrames + TB - 1) / TB;
  const int gOla    = (GL_B * GL_L + TB - 1) / TB;
  const int gBasisF = (GL_FP * GL_NF + TB - 1) / TB;
  const int gWsq    = (GL_SIG + TB - 1) / TB;
  const int gFull   = (GL_B * 128 * 512 + TB - 1) / TB;
  const int gOut    = (GL_B * GL_NS + TB - 1) / TB;
  // GEMM wave tiling: 32x32 C per wave, 8 waves / 256-thread block
  const int gStft   = ((GL_M / 32) * (GL_FP / 32) + 7) / 8;   // 2176 waves -> 272 blocks
  const int gIstft  = ((GL_M / 32) * (GL_NF / 32) + 7) / 8;   // 4096 waves -> 512 blocks

  // constants
  gl_fill_fwd_basis<<<gBasisF, TB, 0, stream>>>(BcT, BsT);
  gl_fill_inv_basis<<<gBasisF, TB, 0, stream>>>(CtT, StT);
  gl_fill_wsqinv  <<<gWsq,    TB, 0, stream>>>(wsqInv);

  // spectrogram prep
  gl_resize_full<<<gFull, TB, 0, stream>>>(params, full_spec);
  gl_make_mag   <<<gSpec, TB, 0, stream>>>(full_spec, mag);
  gl_init_angles<<<gSpec, TB, 0, stream>>>(mag, Sre, Sim, tre, tim);

  // Griffin-Lim iterations
  for (int it = 0; it < 32; ++it) {
    gl_istft_gemm<<<gIstft, TB, 0, stream>>>(Sre, Sim, CtT, StT, ft);
    gl_ola       <<<gOla,   TB, 0, stream>>>(ft, wsqInv, xsig);
    gl_frame     <<<gFrames,TB, 0, stream>>>(xsig, ff);
    gl_stft_gemm <<<gStft,  TB, 0, stream>>>(ff, BcT, BsT, Re, Im);
    gl_update    <<<gSpec,  TB, 0, stream>>>(Re, Im, mag, tre, tim, Sre, Sim);
  }

  // final ISTFT + normalize
  gl_istft_gemm<<<gIstft, TB, 0, stream>>>(Sre, Sim, CtT, StT, ft);
  gl_ola       <<<gOla,   TB, 0, stream>>>(ft, wsqInv, xsig);
  gl_maxabs    <<<GL_B,   TB, 0, stream>>>(xsig, mv);
  gl_scale_out <<<gOut,   TB, 0, stream>>>(xsig, mv, audio_out);
}
